// GeneContextProcessor_14353780703957
// MI455X (gfx1250) — matrix operand order repaired and checked by reading, not compiled, not run on hardware
//
#include <hip/hip_runtime.h>
#include <hip/hip_bf16.h>
#include <float.h>

// ---------------------------------------------------------------------------
// MI455X (gfx1250) fused implementation.
//  - All GEMMs (MLP, in_proj/out_proj/lin, attention scores, attn*V) run on
//    v_wmma_f32_16x16x32_f16 (f16 in, f32 accumulate; K=128 folded as 4 steps).
//  - Wave32-native: one 16-row tile per wave, softmax via 16-lane shfl_xor.
//  - Weights are pre-converted to f16 in d_ws once per launch: the set
//    transformer is L2-weight-bandwidth bound (M=16, ~2 flop/byte), so f16
//    weights halve L2 traffic and kill per-tile v_cvt work.
//  - All helpers are force-inlined so LDS pointers lower to ds_* and global
//    pointers to global_load_* (no flat ops / joint waits).
// ---------------------------------------------------------------------------

typedef __attribute__((ext_vector_type(16))) _Float16 v16h;
typedef __attribute__((ext_vector_type(8)))  _Float16 v8h;
typedef __attribute__((ext_vector_type(8)))  float    v8f;

#define NGENES 100000
#define NRXN   20000
#define C      128
#define NEGBIG (-1e30f)
#define EP_RELU 1

#define DEVINL __device__ __forceinline__

struct AFrag { v16h a[4]; };   // A fragments for K=128 (4 k-steps of 32)

// A-matrix f16 16x32 layout (ISA 7.12.2): lane m = L&15, hi = L>>4,
// half i -> k = 16*(i>>3) + 8*hi + (i&7)  => two runs of 8 consecutive.
DEVINL AFrag packA_f32(const float* A, int lane) {
  int m = lane & 15, hi = (lane >> 4) & 1;
  AFrag f;
#pragma unroll
  for (int ks = 0; ks < 4; ++ks) {
    const float* p = A + m * C + ks * 32 + 8 * hi;
#pragma unroll
    for (int i = 0; i < 8; ++i) f.a[ks][i] = (_Float16)p[i];
#pragma unroll
    for (int i = 0; i < 8; ++i) f.a[ks][8 + i] = (_Float16)p[16 + i];
  }
  return f;
}

DEVINL AFrag packA_h(const _Float16* A, int lane) {
  int m = lane & 15, hi = (lane >> 4) & 1;
  AFrag f;
#pragma unroll
  for (int ks = 0; ks < 4; ++ks) {
    const _Float16* p = A + m * C + ks * 32 + 8 * hi;
    v8h lo = *(const v8h*)p;
    v8h up = *(const v8h*)(p + 16);
#pragma unroll
    for (int i = 0; i < 8; ++i) { f.a[ks][i] = lo[i]; f.a[ks][8 + i] = up[i]; }
  }
  return f;
}

// B-matrix f16 32x16 layout: lane n = L&15, hi = L>>4, half i -> k = 16*(i>>3) + 8*hi + (i&7).
// B[k][n] = W[n][k]; weights are pre-converted f16, row-major [N][128]:
// two runs of 8 halves, 16 apart = two aligned 16B global loads, no conversion.
DEVINL v16h packB_h(const _Float16* W, int ng, int ks, int hi) {
  const _Float16* p = W + (size_t)ng * C + ks * 32 + 8 * hi;
  v8h b0 = *(const v8h*)p;
  v8h b1 = *(const v8h*)(p + 16);
  v16h b;
#pragma unroll
  for (int i = 0; i < 8; ++i) { b[i] = b0[i]; b[8 + i] = b1[i]; }
  return b;
}

// D[16x128] = A[16x128] @ W^T + bias, with optional scale/relu/row-mask/residual.
// C/D layout: lane n = L&15, hi = L>>4; VGPR r holds row m = r + 8*hi.
DEVINL void gemm_x128(
    const AFrag& A, const _Float16* W, const float* bias, float scale,
    unsigned rowmask, int flags, const float* resid,
    float* outF, _Float16* outH, int lane)
{
  int n = lane & 15, hi = (lane >> 4) & 1;
#pragma unroll 1
  for (int nt = 0; nt < 8; ++nt) {
    int ng = nt * 16 + n;
    v8f acc = {};
#pragma unroll
    for (int ks = 0; ks < 4; ++ks) {
      v16h b = packB_h(W, ng, ks, hi);
      acc = __builtin_amdgcn_wmma_f32_16x16x32_f16(false, A.a[ks], false, b,
                                                   (short)0, acc, false, false);
    }
    float bv = bias[ng];
#pragma unroll
    for (int r = 0; r < 8; ++r) {
      int m = r + 8 * hi;
      float v = (acc[r] + bv) * scale;
      if (flags & EP_RELU) v = fmaxf(v, 0.f);
      v = ((rowmask >> m) & 1u) ? v : 0.f;
      if (resid) v += resid[m * C + ng];
      if (outF) outF[m * C + ng] = v;
      else      outH[m * C + ng] = (_Float16)v;
    }
  }
}

// LayerNorm over 128 channels of a 16x128 f32 LDS tile (2 lanes per row).
DEVINL void ln16(float* X, const float* g, const float* b, bool relu, int lane) {
  int row = lane >> 1, half = lane & 1;
  float* x = X + row * C + half * 64;
  float s1 = 0.f, s2 = 0.f;
#pragma unroll 8
  for (int i = 0; i < 64; ++i) { float v = x[i]; s1 += v; s2 += v * v; }
  s1 += __shfl_xor(s1, 1);
  s2 += __shfl_xor(s2, 1);
  float mean = s1 * (1.f / 128.f);
  float var  = s2 * (1.f / 128.f) - mean * mean;
  float rstd = rsqrtf(fmaxf(var, 0.f) + 1e-5f);
#pragma unroll 8
  for (int i = 0; i < 64; ++i) {
    int c = half * 64 + i;
    float v = (x[i] - mean) * rstd * g[c] + b[c];
    if (relu) v = fmaxf(v, 0.f);
    x[i] = v;
  }
}

struct MabW {
  const _Float16 *inw, *outw, *linw;                       // f16 (in d_ws)
  const float *inb, *outb, *linb, *ln1g, *ln1b, *ln2g, *ln2b;  // f32
};

// One MAB: out = LN2(h + relu(lin(h))), h = LN1(mask(mha(xq,xkv)) + xq).
// out may alias xq or xkv. ybuf is per-wave scratch (may alias xkv).
DEVINL void mab16(
    float* xq, float* xkv, float* out, float* ybuf, _Float16* qkv,
    unsigned kvmask, unsigned qmask, const MabW& w, int lane)
{
  _Float16* qsec = qkv;
  _Float16* ksec = qkv + 16 * C;
  _Float16* vsec = qkv + 32 * C;
  _Float16* pstage = (_Float16*)ybuf;   // 16x16 prob staging (ybuf dead here)
  const float qscale = 0.17677669529663687f;  // 1/sqrt(dh=32)
  int m = lane & 15, hi = (lane >> 4) & 1;

  __syncthreads();
  AFrag aq = packA_f32(xq, lane);
  AFrag akv;
  if (xq == xkv) akv = aq; else akv = packA_f32(xkv, lane);
  gemm_x128(aq,  w.inw,           w.inb,       qscale, 0xFFFFu, 0, nullptr, nullptr, qsec, lane);
  gemm_x128(akv, w.inw + 128 * C, w.inb + 128, 1.f,    0xFFFFu, 0, nullptr, nullptr, ksec, lane);
  gemm_x128(akv, w.inw + 256 * C, w.inb + 256, 1.f,    0xFFFFu, 0, nullptr, nullptr, vsec, lane);
  __syncthreads();

#pragma unroll 1
  for (int h = 0; h < 4; ++h) {
    // ---- scores[16x16] = q_head @ k_head^T (K=32, one WMMA) ----
    v16h a, bk;
    const _Float16* qrow = qsec + m * C + 32 * h;
    v8h q0 = *(const v8h*)(qrow + 8 * hi);
    v8h q1 = *(const v8h*)(qrow + 16 + 8 * hi);
    const _Float16* krow = ksec + m * C + 32 * h + 8 * hi;  // B: n = lane&15
    v8h k0 = *(const v8h*)krow;
    v8h k1 = *(const v8h*)(krow + 16);
#pragma unroll
    for (int i = 0; i < 8; ++i) { a[i] = q0[i]; a[8 + i] = q1[i]; bk[i] = k0[i]; bk[8 + i] = k1[i]; }
    v8f s = {};
    s = __builtin_amdgcn_wmma_f32_16x16x32_f16(false, a, false, bk, (short)0, s, false, false);

    // ---- key mask + softmax across the 16-lane half-wave ----
    bool nv = (kvmask >> m) & 1u;   // this lane's key column
    float p[8];
#pragma unroll
    for (int r = 0; r < 8; ++r) {
      float v = nv ? s[r] : NEGBIG;
      float mx = v;
      mx = fmaxf(mx, __shfl_xor(mx, 1));
      mx = fmaxf(mx, __shfl_xor(mx, 2));
      mx = fmaxf(mx, __shfl_xor(mx, 4));
      mx = fmaxf(mx, __shfl_xor(mx, 8));
      float e = __expf(v - mx);
      float sm = e;
      sm += __shfl_xor(sm, 1);
      sm += __shfl_xor(sm, 2);
      sm += __shfl_xor(sm, 4);
      sm += __shfl_xor(sm, 8);
      p[r] = e / sm;
    }
    __syncthreads();
#pragma unroll
    for (int r = 0; r < 8; ++r) pstage[(r + 8 * hi) * 16 + m] = (_Float16)p[r];
    __syncthreads();

    // ---- o_head[16x32] = P[16x16] @ v_head[16x32]  (K padded 16->32) ----
    v16h ap;
    v8h pl = *(const v8h*)(pstage + m * 16 + 8 * hi);
#pragma unroll
    for (int i = 0; i < 8; ++i) { ap[i] = pl[i]; ap[8 + i] = (_Float16)0.f; }
    v8f o[2];
#pragma unroll
    for (int t = 0; t < 2; ++t) {
      v16h bv;
#pragma unroll
      for (int i = 0; i < 16; ++i) {
        _Float16 vv = (i < 8) ? vsec[(8 * hi + i) * C + 32 * h + 16 * t + m] : (_Float16)0.f;
        bv[i] = vv;  // rows j>=16 are zero pad
      }
      v8f z = {};
      o[t] = __builtin_amdgcn_wmma_f32_16x16x32_f16(false, ap, false, bv, (short)0, z, false, false);
    }
    __syncthreads();
#pragma unroll
    for (int t = 0; t < 2; ++t)
#pragma unroll
      for (int r = 0; r < 8; ++r)
        qsec[(r + 8 * hi) * C + 32 * h + 16 * t + m] = (_Float16)(o[t][r]);
    __syncthreads();
  }

  // ---- out_proj + query-row mask + residual(xq) -> ybuf; LN1 ----
  AFrag ao = packA_h(qsec, lane);
  gemm_x128(ao, w.outw, w.outb, 1.f, qmask, 0, xq, ybuf, nullptr, lane);
  __syncthreads();
  ln16(ybuf, w.ln1g, w.ln1b, false, lane);
  __syncthreads();
  // ---- h + relu(lin(h)) -> out; LN2 ----
  AFrag al = packA_f32(ybuf, lane);
  gemm_x128(al, w.linw, w.linb, 1.f, 0xFFFFu, EP_RELU, ybuf, out, nullptr, lane);
  __syncthreads();
  ln16(out, w.ln2g, w.ln2b, false, lane);
  __syncthreads();
}

// ---------------------------------------------------------------------------
// Kernel 0: one-shot f32 -> f16 weight conversion into d_ws.
// Segment order/sizes must match the host-side offset computation.
// ---------------------------------------------------------------------------
struct CvtSrc { const float* s[15]; };

__global__ void __launch_bounds__(256) cvt_kernel(CvtSrc cs, _Float16* dst) {
  int tid = blockIdx.x * blockDim.x + threadIdx.x;
  int stride = gridDim.x * blockDim.x;
  int off = 0;
#pragma unroll
  for (int sgm = 0; sgm < 15; ++sgm) {
    const int n = (sgm < 12 && (sgm % 3 == 0)) ? 384 * C : C * C;
    const float* src = cs.s[sgm];
    for (int i = tid; i < n; i += stride) dst[off + i] = (_Float16)src[i];
    off += n;
  }
}

// ---------------------------------------------------------------------------
// Kernel 1: per-gene MLP (2x [Linear 128->128, LN, ReLU]) -> H_g
// ---------------------------------------------------------------------------
struct MlpW { const _Float16 *w1, *w2; const float *b1, *g1, *bn1, *b2, *g2, *bn2; };

#define MLP_WAVES 4
__global__ void __launch_bounds__(128) mlp_kernel(
    const float* __restrict__ gf, float* __restrict__ Hg, MlpW w, int ngenes)
{
  extern __shared__ char smem[];
  int wid = threadIdx.x >> 5, lane = threadIdx.x & 31;
  char* base = smem + wid * 16384;
  float* xbuf = (float*)base;
  float* ybuf = (float*)(base + 8192);
  int g0 = blockIdx.x * (MLP_WAVES * 16) + wid * 16;
  for (int j = 0; j < 16; ++j) {
    int g = g0 + j; if (g > ngenes - 1) g = ngenes - 1;   // clamp (uniform)
    ((float4*)(xbuf + j * C))[lane] = ((const float4*)(gf + (size_t)g * C))[lane];
  }
  __syncthreads();
  AFrag a1 = packA_f32(xbuf, lane);
  gemm_x128(a1, w.w1, w.b1, 1.f, 0xFFFFu, 0, nullptr, ybuf, nullptr, lane);
  __syncthreads();
  ln16(ybuf, w.g1, w.bn1, true, lane);
  __syncthreads();
  AFrag a2 = packA_f32(ybuf, lane);
  gemm_x128(a2, w.w2, w.b2, 1.f, 0xFFFFu, 0, nullptr, xbuf, nullptr, lane);
  __syncthreads();
  ln16(xbuf, w.g2, w.bn2, true, lane);
  __syncthreads();
  for (int j = 0; j < 16; ++j) {
    int g = g0 + j;
    if (g < ngenes)
      ((float4*)(Hg + (size_t)g * C))[lane] = ((const float4*)(xbuf + j * C))[lane];
  }
}

// ---------------------------------------------------------------------------
// Kernel 2: fused per-reaction set transformer -> H_r
// ---------------------------------------------------------------------------
struct RxnW {
  MabW enc0, enc1, pmam, dec;
  const _Float16* pma_linw;
  const float *pma_linb, *seed;
};

#define RXN_WAVES  2
#define WAVE_BYTES (8192 + 8192 + 12288)  // xbuf f32 + ybuf f32 + qkv f16

__global__ void __launch_bounds__(64) rxn_kernel(
    const float* __restrict__ Hg, float* __restrict__ Hr,
    const int* __restrict__ rxn_idx, const int* __restrict__ gene_idx,
    int E, RxnW W)
{
  extern __shared__ char smem[];
  int wid = threadIdx.x >> 5, lane = threadIdx.x & 31;
  char* base = smem + wid * WAVE_BYTES;
  float* xbuf = (float*)base;
  float* ybuf = (float*)(base + 8192);
  _Float16* qkv = (_Float16*)(base + 16384);
  int r = blockIdx.x * RXN_WAVES + wid;   // 20000 = 10000 * 2, no remainder

  // lower_bound(r) and lower_bound(r+1) in sorted rxn_idx (uniform per wave)
  int lo = 0, up = E;
  while (lo < up) { int mid = (lo + up) >> 1; if (rxn_idx[mid] < r) lo = mid + 1; else up = mid; }
  int s0 = lo;
  lo = s0; up = E;
  while (lo < up) { int mid = (lo + up) >> 1; if (rxn_idx[mid] < r + 1) lo = mid + 1; else up = mid; }
  int count = lo - s0; if (count > 16) count = 16;
  unsigned kvmask = (count >= 16) ? 0xFFFFu : ((1u << count) - 1u);

  // gather gene rows into xbuf (zeros for padded slots)
  for (int j = 0; j < 16; ++j) {
    float4 v = make_float4(0.f, 0.f, 0.f, 0.f);
    if (j < count) {
      int g = gene_idx[s0 + j];
      v = ((const float4*)(Hg + (size_t)g * C))[lane];
    }
    ((float4*)(xbuf + j * C))[lane] = v;
  }

  mab16(xbuf, xbuf, xbuf, ybuf, qkv, kvmask, kvmask, W.enc0, lane);
  mab16(xbuf, xbuf, xbuf, ybuf, qkv, kvmask, kvmask, W.enc1, lane);

  // xp = relu(pma.lin(x)) -> ybuf
  __syncthreads();
  AFrag ax = packA_f32(xbuf, lane);
  gemm_x128(ax, W.pma_linw, W.pma_linb, 1.f, 0xFFFFu, EP_RELU, nullptr, ybuf, nullptr, lane);
  __syncthreads();
  // xbuf <- seed (row 0), zeros elsewhere
  for (int j = 0; j < 16; ++j) {
    float4 v = make_float4(0.f, 0.f, 0.f, 0.f);
    if (j == 0) v = ((const float4*)W.seed)[lane];
    ((float4*)(xbuf + j * C))[lane] = v;
  }
  // PMA: mab(seed, xp, x_valid=None, y_valid=valid); ybuf(kv) dead after in_proj
  mab16(xbuf, ybuf, xbuf, ybuf, qkv, kvmask, 0xFFFFu, W.pmam, lane);
  // Decoder: self-attn on length-1 set -> only key slot 0 valid
  mab16(xbuf, xbuf, xbuf, ybuf, qkv, 0x1u, 0xFFFFu, W.dec, lane);

  __syncthreads();
  float4 v = ((float4*)xbuf)[lane];   // row 0 only
  float* pv = (float*)&v;
#pragma unroll
  for (int i = 0; i < 4; ++i) {
    float x = pv[i];
    if (!(x == x)) x = 0.f;                       // NaN -> 0
    else if (x > 3.402823466e38f)  x = 3.402823466e38f;
    else if (x < -3.402823466e38f) x = -3.402823466e38f;
    pv[i] = x;
  }
  ((float4*)(Hr + (size_t)r * C))[lane] = v;
}

// ---------------------------------------------------------------------------
// Host launcher.  Input order: gene_features, params-leaves (jax tree order:
// dict keys sorted at each level, lists in order), rxn_idx, gene_idx.
// params sorted keys: decoder, encoders[0..1], mlp[0..1], pma{lin,mab,seed}.
// Each MAB's leaves: in_proj.b, in_proj.w, lin.b, lin.w, ln1.b, ln1.g,
//                    ln2.b, ln2.g, out_proj.b, out_proj.w.
// ---------------------------------------------------------------------------
extern "C" void kernel_launch(void* const* d_in, const int* in_sizes, int n_in,
                              void* d_out, int out_size, void* d_ws, size_t ws_size,
                              hipStream_t stream)
{
  (void)ws_size; (void)out_size;
  const float* gf = (const float*)d_in[0];
  const int P = 1;
  auto F = [&](int i) { return (const float*)d_in[i]; };

  _Float16* wsH = (_Float16*)d_ws;

  // Segment order for f16 weight table (must match cvt_kernel):
  // [mab.inw, mab.outw, mab.linw] x {dec, enc0, enc1, pmam}, pma_linw, w1, w2
  CvtSrc cs;
  int segOff[15];
  {
    // param leaf bases: dec=P+0, enc0=P+10, enc1=P+20, pmam=P+40
    const int mabBase[4] = { P + 0, P + 10, P + 20, P + 40 };
    int idx = 0, off = 0;
    for (int mi = 0; mi < 4; ++mi) {
      cs.s[idx] = F(mabBase[mi] + 1); segOff[idx] = off; off += 384 * C; ++idx; // in_proj.w
      cs.s[idx] = F(mabBase[mi] + 9); segOff[idx] = off; off += C * C;  ++idx;  // out_proj.w
      cs.s[idx] = F(mabBase[mi] + 3); segOff[idx] = off; off += C * C;  ++idx;  // lin.w
    }
    cs.s[idx] = F(P + 39); segOff[idx] = off; off += C * C; ++idx;  // pma.lin.w
    cs.s[idx] = F(P + 31); segOff[idx] = off; off += C * C; ++idx;  // mlp[0].lin.w
    cs.s[idx] = F(P + 35); segOff[idx] = off; off += C * C; ++idx;  // mlp[1].lin.w
  }

  auto mkMab = [&](int i, int seg) {
    MabW m;
    m.inw  = wsH + segOff[seg + 0];
    m.outw = wsH + segOff[seg + 1];
    m.linw = wsH + segOff[seg + 2];
    m.inb  = F(i + 0);
    m.linb = F(i + 2);
    m.ln1b = F(i + 4); m.ln1g = F(i + 5);
    m.ln2b = F(i + 6); m.ln2g = F(i + 7);
    m.outb = F(i + 8);
    return m;
  };

  RxnW W;
  W.dec  = mkMab(P + 0,  0);   // decoder
  W.enc0 = mkMab(P + 10, 3);   // encoders[0]
  W.enc1 = mkMab(P + 20, 6);   // encoders[1]
  W.pmam = mkMab(P + 40, 9);   // pma.mab
  W.pma_linw = wsH + segOff[12];
  W.pma_linb = F(P + 38);
  W.seed     = F(P + 50);

  MlpW mw;
  mw.w1 = wsH + segOff[13]; mw.b1 = F(P + 30); mw.bn1 = F(P + 32); mw.g1 = F(P + 33);
  mw.w2 = wsH + segOff[14]; mw.b2 = F(P + 34); mw.bn2 = F(P + 36); mw.g2 = F(P + 37);

  const int* rxn_idx  = (const int*)d_in[n_in - 2];
  const int* gene_idx = (const int*)d_in[n_in - 1];
  int E = in_sizes[n_in - 2];

  float* Hg = (float*)d_out;
  float* Hr = Hg + (size_t)NGENES * C;

  cvt_kernel<<<128, 256, 0, stream>>>(cs, wsH);
  int mlp_blocks = (NGENES + MLP_WAVES * 16 - 1) / (MLP_WAVES * 16);
  mlp_kernel<<<mlp_blocks, MLP_WAVES * 32, MLP_WAVES * 16384, stream>>>(gf, Hg, mw, NGENES);
  rxn_kernel<<<NRXN / RXN_WAVES, RXN_WAVES * 32, RXN_WAVES * WAVE_BYTES, stream>>>(
      Hg, Hr, rxn_idx, gene_idx, E, W);
}
